// FEMREncoderLayer_36524401885910
// MI455X (gfx1250) — matrix operand order repaired and checked by reading, not compiled, not run on hardware
//
#include <hip/hip_runtime.h>
#include <hip/hip_bf16.h>

typedef __attribute__((ext_vector_type(16))) _Float16 v16h;
typedef __attribute__((ext_vector_type(8)))  _Float16 v8h;
typedef __attribute__((ext_vector_type(8)))  float    v8f;
typedef int v4i __attribute__((vector_size(16)));

#define HIDDEN 768
#define FF     3072
#define NH     12
#define HD     64
#define NB     2
#define SS     4096
#define ROWS   (NB * SS)          // 8192
#define TOUT   (3 * HIDDEN + FF)  // 5376
#define QKV    (3 * HIDDEN)       // 2304
#define WIN    512

#if defined(__has_builtin)
#  if __has_builtin(__builtin_amdgcn_global_load_async_to_lds_b128) && \
      __has_builtin(__builtin_amdgcn_s_wait_asynccnt)
#    define USE_ASYNC 1
#  endif
#endif
#ifndef USE_ASYNC
#  define USE_ASYNC 0
#endif

// ---------------------------------------------------------------- cast f32->f16 (8/thread)
__global__ __launch_bounds__(256) void cast_f16_kernel(const float4* __restrict__ s,
                                                       v8h* __restrict__ d, long n8) {
  long i = (long)blockIdx.x * 256 + threadIdx.x;
  if (i < n8) {
    float4 a = s[i * 2], b = s[i * 2 + 1];
    v8h o;
    o[0] = (_Float16)a.x; o[1] = (_Float16)a.y; o[2] = (_Float16)a.z; o[3] = (_Float16)a.w;
    o[4] = (_Float16)b.x; o[5] = (_Float16)b.y; o[6] = (_Float16)b.z; o[7] = (_Float16)b.w;
    d[i] = o;
  }
}

// ---------------------------------------------------------------- RMSNorm -> f16
__global__ __launch_bounds__(256) void rmsnorm_kernel(const float* __restrict__ x,
                                                      const float* __restrict__ w,
                                                      _Float16* __restrict__ xn) {
  int row = blockIdx.x;
  const float* xr = x + (long)row * HIDDEN;
  float v[3];
  float s = 0.f;
#pragma unroll
  for (int i = 0; i < 3; ++i) {
    v[i] = xr[threadIdx.x + 256 * i];
    s += v[i] * v[i];
  }
#pragma unroll
  for (int m = 16; m >= 1; m >>= 1) s += __shfl_xor(s, m, 32);
  __shared__ float red[8];
  int wave = threadIdx.x >> 5;
  if ((threadIdx.x & 31) == 0) red[wave] = s;
  __syncthreads();
  float tot = 0.f;
#pragma unroll
  for (int i = 0; i < 8; ++i) tot += red[i];
  float r = rsqrtf(tot * (1.0f / HIDDEN) + 1e-6f);
#pragma unroll
  for (int i = 0; i < 3; ++i) {
    int c = threadIdx.x + 256 * i;
    xn[(long)row * HIDDEN + c] = (_Float16)(v[i] * r * w[c]);
  }
}

// ---------------------------------------------------------------- WMMA GEMM
// C[M,N] = A[M,K] * B[K,N] + bias.  A f16 row-major (optionally split at
// k=ksplit into A/A2), B f16 row-major.  Output f32 (outF) or f16 with exact
// GeLU applied to columns >= gelu_start (outH).
#define TM 128
#define TN 128
#define TK 32
#define LSTR 56   // padded LDS row stride in halves (112B: 16B aligned)

__global__ __launch_bounds__(256) void gemm_wmma_kernel(
    const _Float16* __restrict__ A, int lda,
    const _Float16* __restrict__ A2, int lda2, int ksplit, int a2off,
    const _Float16* __restrict__ Bw,
    const float* __restrict__ bias,
    float* __restrict__ outF, _Float16* __restrict__ outH,
    int gelu_start, int M, int N, int K) {
  __shared__ _Float16 As[TM][LSTR];   // [row][k]
  __shared__ _Float16 Bs[TN][LSTR];   // transposed: [n][k]

  const int row0 = blockIdx.y * TM;
  const int col0 = blockIdx.x * TN;
  const int tid  = threadIdx.x;
  const int wave = tid >> 5, lane = tid & 31;
  const int wm = wave & 1;        // 0..1 -> 64-row half
  const int wn = wave >> 1;       // 0..3 -> 32-col quarter
  const int m16 = lane & 15;
  const int hiH = lane >> 4;
  const int kbA = hiH ? 8 : 0;
  const int kbB = hiH ? 16 : 0;

  v8f acc[4][2] = {};

  for (int kk = 0; kk < K; kk += TK) {
    // ---- A tile 128x32: 512 b128 chunks, 2 per thread, async copy to LDS
#pragma unroll
    for (int i = 0; i < 2; ++i) {
      int cid = i * 256 + tid;          // chunk id
      int r = cid >> 2, off = (cid & 3) * 8;
      int kg = kk + off;                // chunk never straddles ksplit (32-aligned)
      const _Float16* gsrc;
      if (!A2 || kg < ksplit)
        gsrc = A + (long)(row0 + r) * lda + kg;
      else
        gsrc = A2 + (long)(row0 + r) * lda2 + a2off + (kg - ksplit);
#if USE_ASYNC
      __builtin_amdgcn_global_load_async_to_lds_b128((v4i*)gsrc, (v4i*)&As[r][off], 0, 0);
#else
      *(v8h*)&As[r][off] = *(const v8h*)gsrc;
#endif
    }
    // ---- B tile 32x128: vectorized global load, transposed store into LDS
#pragma unroll
    for (int i = 0; i < 2; ++i) {
      int cid = i * 256 + tid;
      int r = cid >> 4, c8 = (cid & 15) * 8;
      v8h bvv = *(const v8h*)&Bw[(long)(kk + r) * N + col0 + c8];
#pragma unroll
      for (int t = 0; t < 8; ++t) Bs[c8 + t][r] = bvv[t];
    }
    if (kk + TK < K)
      __builtin_prefetch(&Bw[(long)(kk + TK) * N + col0 + (tid & 127) * 8], 0, 1);
#if USE_ASYNC
    __builtin_amdgcn_s_wait_asynccnt(0);
#endif
    __syncthreads();

    v16h a[4], b[2];
#pragma unroll
    for (int i = 0; i < 4; ++i) {
      const _Float16* p = &As[wm * 64 + i * 16 + m16][kbA];
      v8h lo = *(const v8h*)p;
      v8h hi = *(const v8h*)(p + 16);
#pragma unroll
      for (int t = 0; t < 8; ++t) { a[i][t] = lo[t]; a[i][t + 8] = hi[t]; }
    }
#pragma unroll
    for (int j = 0; j < 2; ++j) {
      const _Float16* q = &Bs[wn * 32 + j * 16 + m16][kbA];
      v8h lo = *(const v8h*)q;
      v8h hi = *(const v8h*)(q + 16);
#pragma unroll
      for (int t = 0; t < 8; ++t) { b[j][t] = lo[t]; b[j][t + 8] = hi[t]; }
    }

#pragma unroll
    for (int i = 0; i < 4; ++i)
#pragma unroll
      for (int j = 0; j < 2; ++j)
        acc[i][j] = __builtin_amdgcn_wmma_f32_16x16x32_f16(
            false, a[i], false, b[j], (short)0, acc[i][j], false, false);
    __syncthreads();
  }

  // ---- epilogue
#pragma unroll
  for (int i = 0; i < 4; ++i)
#pragma unroll
    for (int j = 0; j < 2; ++j) {
      int c = col0 + wn * 32 + j * 16 + m16;
      float bv = bias[c];
#pragma unroll
      for (int v = 0; v < 8; ++v) {
        int r = row0 + wm * 64 + i * 16 + v + hiH * 8;
        float val = acc[i][j][v] + bv;
        if (outH) {
          if (c >= gelu_start)
            val = 0.5f * val * (1.0f + erff(val * 0.70710678118f));
          outH[(long)r * N + c] = (_Float16)val;
        } else {
          outF[(long)r * N + c] = val;
        }
      }
    }
}

// ---------------------------------------------------------------- RoPE + V transpose
// One thread per (row, head, 16-wide d-quarter): vectorized loads/stores.
__global__ __launch_bounds__(256) void rope_kernel(const _Float16* __restrict__ t,
                                                   const float* __restrict__ sinp,
                                                   const float* __restrict__ cosp,
                                                   _Float16* __restrict__ qh,
                                                   _Float16* __restrict__ kh,
                                                   _Float16* __restrict__ vT) {
  long idx = (long)blockIdx.x * 256 + threadIdx.x;   // ROWS*NH*4 total
  int qd = (int)(idx & 3);
  long t1 = idx >> 2;
  int h = (int)(t1 % NH);
  long row = t1 / NH;
  int s = (int)(row & (SS - 1));
  int b = (int)(row >> 12);
  int db = qd * 16;

  const _Float16* tr = t + row * TOUT + h * HD + db;
  v8h q8[2], k8[2], v8[2];
#pragma unroll
  for (int i = 0; i < 2; ++i) {
    q8[i] = *(const v8h*)(tr + 0 * HIDDEN + i * 8);
    k8[i] = *(const v8h*)(tr + 1 * HIDDEN + i * 8);
    v8[i] = *(const v8h*)(tr + 2 * HIDDEN + i * 8);
  }
  float sn[16], cs[16];
  const float4* sp = (const float4*)(sinp + (long)s * HD + db);
  const float4* cp = (const float4*)(cosp + (long)s * HD + db);
#pragma unroll
  for (int i = 0; i < 4; ++i) {
    float4 a = sp[i]; float4 c = cp[i];
    sn[i * 4 + 0] = a.x; sn[i * 4 + 1] = a.y; sn[i * 4 + 2] = a.z; sn[i * 4 + 3] = a.w;
    cs[i * 4 + 0] = c.x; cs[i * 4 + 1] = c.y; cs[i * 4 + 2] = c.z; cs[i * 4 + 3] = c.w;
  }
  v8h qo[2], ko[2];
#pragma unroll
  for (int p = 0; p < 8; ++p) {
    int d0 = p * 2, d1 = p * 2 + 1;
    float q0 = (float)q8[d0 >> 3][d0 & 7], q1 = (float)q8[d1 >> 3][d1 & 7];
    float k0 = (float)k8[d0 >> 3][d0 & 7], k1 = (float)k8[d1 >> 3][d1 & 7];
    qo[d0 >> 3][d0 & 7] = (_Float16)(q0 * cs[d0] - q1 * sn[d0]);
    qo[d1 >> 3][d1 & 7] = (_Float16)(q1 * cs[d1] + q0 * sn[d1]);
    ko[d0 >> 3][d0 & 7] = (_Float16)(k0 * cs[d0] - k1 * sn[d0]);
    ko[d1 >> 3][d1 & 7] = (_Float16)(k1 * cs[d1] + k0 * sn[d1]);
  }
  long o = row * HIDDEN + h * HD + db;
  *(v8h*)(qh + o) = qo[0];
  *(v8h*)(qh + o + 8) = qo[1];
  *(v8h*)(kh + o) = ko[0];
  *(v8h*)(kh + o + 8) = ko[1];
  long vb = (((long)b * NH + h) * HD + db) * SS + s;
#pragma unroll
  for (int i = 0; i < 16; ++i)
    vT[vb + (long)i * SS] = v8[i >> 3][i & 7];
}

// ---------------------------------------------------------------- flash-style windowed attention
__global__ __launch_bounds__(256) void attn_kernel(const _Float16* __restrict__ qh,
                                                   const _Float16* __restrict__ kh,
                                                   const _Float16* __restrict__ vT,
                                                   _Float16* __restrict__ attn) {
  __shared__ _Float16 plds[8][16][48];   // per-wave P staging

  const int wave = threadIdx.x >> 5, lane = threadIdx.x & 31;
  const int idx = blockIdx.x * 8 + wave;       // 0..6143
  const int qt = idx & 255;
  const int t1 = idx >> 8;
  const int h = t1 % NH;
  const int b = t1 / NH;
  const int q0s = qt * 16;
  const long rowbase = (long)b * SS;
  const int m16 = lane & 15, hiH = lane >> 4;
  const int kbA = hiH ? 8 : 0;
  const int kbB = hiH ? 16 : 0;

  v16h aq[2];
#pragma unroll
  for (int g = 0; g < 2; ++g) {
    const _Float16* p = qh + (rowbase + q0s + m16) * HIDDEN + h * HD + g * 32 + kbA;
    v8h lo = *(const v8h*)p;
    v8h hi = *(const v8h*)(p + 16);
#pragma unroll
    for (int t = 0; t < 8; ++t) { aq[g][t] = lo[t]; aq[g][t + 8] = hi[t]; }
  }

  v8f o[4] = {};
  float mrun[8], lrun[8];
#pragma unroll
  for (int v = 0; v < 8; ++v) { mrun[v] = -1e30f; lrun[v] = 0.f; }

  int start = q0s - (WIN - 1);
  if (start < 0) start = 0;
  start &= ~31;
  const int endk = q0s + 15;
  const float scale = 0.125f;   // 1/sqrt(64)
  const long vbase = ((long)b * NH + h) * HD;

  for (int ck = start; ck <= endk; ck += 32) {
    v8f sc[2] = {};
#pragma unroll
    for (int cf = 0; cf < 2; ++cf) {
      int kpos = ck + cf * 16 + m16;
#pragma unroll
      for (int g = 0; g < 2; ++g) {
        const _Float16* kp = kh + (rowbase + kpos) * HIDDEN + h * HD + g * 32 + kbA;
        v8h klo = *(const v8h*)kp, khi = *(const v8h*)(kp + 16);
        v16h bk;
#pragma unroll
        for (int t = 0; t < 8; ++t) { bk[t] = klo[t]; bk[t + 8] = khi[t]; }
        sc[cf] = __builtin_amdgcn_wmma_f32_16x16x32_f16(
            false, aq[g], false, bk, (short)0, sc[cf], false, false);
      }
    }
#pragma unroll
    for (int cf = 0; cf < 2; ++cf)
#pragma unroll
      for (int v = 0; v < 8; ++v) {
        int gq = q0s + v + hiH * 8;
        int gk = ck + cf * 16 + m16;
        bool ok = (gk <= gq) && (gq - gk < WIN);
        sc[cf][v] = ok ? sc[cf][v] * scale : -1e30f;
      }
    float mnew[8];
#pragma unroll
    for (int v = 0; v < 8; ++v) {
      float mx = fmaxf(sc[0][v], sc[1][v]);
#pragma unroll
      for (int msk = 8; msk >= 1; msk >>= 1) mx = fmaxf(mx, __shfl_xor(mx, msk, 32));
      mnew[v] = fmaxf(mrun[v], mx);
    }
#pragma unroll
    for (int v = 0; v < 8; ++v) {
      float alpha = __expf(mrun[v] - mnew[v]);
      lrun[v] *= alpha;
#pragma unroll
      for (int t = 0; t < 4; ++t) o[t][v] *= alpha;
      float p0 = __expf(sc[0][v] - mnew[v]);
      float p1 = __expf(sc[1][v] - mnew[v]);
      sc[0][v] = p0; sc[1][v] = p1;
      float ps = p0 + p1;
#pragma unroll
      for (int msk = 8; msk >= 1; msk >>= 1) ps += __shfl_xor(ps, msk, 32);
      lrun[v] += ps;
      mrun[v] = mnew[v];
    }
#pragma unroll
    for (int cf = 0; cf < 2; ++cf)
#pragma unroll
      for (int v = 0; v < 8; ++v)
        plds[wave][v + hiH * 8][cf * 16 + m16] = (_Float16)sc[cf][v];
    asm volatile("s_wait_dscnt 0x0" ::: "memory");
    v16h ap;
    {
      const _Float16* p = &plds[wave][m16][kbA];
      v8h lo = *(const v8h*)p;
      v8h hi = *(const v8h*)(p + 16);
#pragma unroll
      for (int t = 0; t < 8; ++t) { ap[t] = lo[t]; ap[t + 8] = hi[t]; }
    }
#pragma unroll
    for (int t = 0; t < 4; ++t) {
      const _Float16* vp = vT + (vbase + t * 16 + m16) * SS + ck + kbA;
      v8h vlo = *(const v8h*)vp, vhi = *(const v8h*)(vp + 16);
      v16h bv;
#pragma unroll
      for (int u = 0; u < 8; ++u) { bv[u] = vlo[u]; bv[u + 8] = vhi[u]; }
      o[t] = __builtin_amdgcn_wmma_f32_16x16x32_f16(
          false, ap, false, bv, (short)0, o[t], false, false);
    }
  }

#pragma unroll
  for (int t = 0; t < 4; ++t)
#pragma unroll
    for (int v = 0; v < 8; ++v) {
      long r = rowbase + q0s + v + hiH * 8;
      attn[r * HIDDEN + h * HD + t * 16 + m16] = (_Float16)(o[t][v] / lrun[v]);
    }
}

// ---------------------------------------------------------------- launch
extern "C" void kernel_launch(void* const* d_in, const int* in_sizes, int n_in,
                              void* d_out, int out_size, void* d_ws, size_t ws_size,
                              hipStream_t stream) {
  const float* x     = (const float*)d_in[0];
  const float* sinp  = (const float*)d_in[1];
  const float* cosp  = (const float*)d_in[2];
  const float* wnorm = (const float*)d_in[3];
  const float* W_in  = (const float*)d_in[4];
  const float* b_in  = (const float*)d_in[5];
  const float* W_out = (const float*)d_in[6];
  const float* b_out = (const float*)d_in[7];
  float* out = (float*)d_out;

  char* wsp = (char*)d_ws;
  auto alloc = [&](size_t bytes) -> char* {
    char* p = wsp;
    wsp += (bytes + 255) & ~(size_t)255;
    return p;
  };
  _Float16* W_in_h  = (_Float16*)alloc((size_t)HIDDEN * TOUT * 2);
  _Float16* W_out_h = (_Float16*)alloc((size_t)(HIDDEN + FF) * HIDDEN * 2);
  _Float16* xn_h    = (_Float16*)alloc((size_t)ROWS * HIDDEN * 2);
  _Float16* t_h     = (_Float16*)alloc((size_t)ROWS * TOUT * 2);
  _Float16* q_h     = (_Float16*)alloc((size_t)ROWS * HIDDEN * 2);
  _Float16* k_h     = (_Float16*)alloc((size_t)ROWS * HIDDEN * 2);
  _Float16* vT      = (_Float16*)alloc((size_t)ROWS * HIDDEN * 2);
  _Float16* attn_h  = (_Float16*)alloc((size_t)ROWS * HIDDEN * 2);

  {
    long n1 = (long)HIDDEN * TOUT / 8;
    cast_f16_kernel<<<(int)((n1 + 255) / 256), 256, 0, stream>>>(
        (const float4*)W_in, (v8h*)W_in_h, n1);
    long n2 = (long)(HIDDEN + FF) * HIDDEN / 8;
    cast_f16_kernel<<<(int)((n2 + 255) / 256), 256, 0, stream>>>(
        (const float4*)W_out, (v8h*)W_out_h, n2);
  }
  rmsnorm_kernel<<<ROWS, 256, 0, stream>>>(x, wnorm, xn_h);
  gemm_wmma_kernel<<<dim3(TOUT / TN, ROWS / TM), 256, 0, stream>>>(
      xn_h, HIDDEN, nullptr, 0, HIDDEN, 0,
      W_in_h, b_in, nullptr, t_h, QKV, ROWS, TOUT, HIDDEN);
  {
    long nthreads = (long)ROWS * NH * 4;
    rope_kernel<<<(int)(nthreads / 256), 256, 0, stream>>>(t_h, sinp, cosp, q_h, k_h, vT);
  }
  attn_kernel<<<(NB * NH * (SS / 16)) / 8, 256, 0, stream>>>(q_h, k_h, vT, attn_h);
  gemm_wmma_kernel<<<dim3(HIDDEN / TN, ROWS / TM), 256, 0, stream>>>(
      attn_h, HIDDEN, t_h, TOUT, HIDDEN, QKV,
      W_out_h, b_out, out, nullptr, 1 << 30, ROWS, HIDDEN, HIDDEN + FF);
}
